// LatentAttention_56942676410946
// MI455X (gfx1250) — matrix operand order repaired and checked by reading, not compiled, not run on hardware
//
#include <hip/hip_runtime.h>
#include <stdint.h>

typedef __bf16 bf16_t;
typedef __attribute__((ext_vector_type(8)))  __bf16 v8bf;
typedef __attribute__((ext_vector_type(16))) __bf16 v16bf;
typedef __attribute__((ext_vector_type(8)))  float  v8f;

#define V8F_ZERO ((v8f){0.f,0.f,0.f,0.f,0.f,0.f,0.f,0.f})

// ---------- helpers ----------

__device__ inline bf16_t f2bf(float f) {
  union { float f; uint32_t u; } v; v.f = f;
  uint32_t r = v.u + 0x7FFFu + ((v.u >> 16) & 1u);   // round-to-nearest-even
  uint16_t h = (uint16_t)(r >> 16);
  bf16_t out;
  __builtin_memcpy(&out, &h, 2);
  return out;
}

// A-matrix 16x32 bf16 fragment (ISA 7.12.2): lane = row (caller adds lane&15),
// half (lane>>4) selects K in {h*8..h*8+7} and {16+h*8..16+h*8+7}.
__device__ inline v16bf load_frag_a(const bf16_t* src, int row, int kbase, int ld) {
  const int lane = threadIdx.x & 31;
  const int half = lane >> 4;
  const bf16_t* p = src + (size_t)row * ld + kbase + half * 8;
  v8bf lo = *(const v8bf*)p;          // K = h*8 .. h*8+7
  v8bf hi = *(const v8bf*)(p + 16);   // K = 16+h*8 .. 16+h*8+7
  return __builtin_shufflevector(lo, hi, 0,1,2,3,4,5,6,7,8,9,10,11,12,13,14,15);
}

// B-matrix 32x16 bf16 fragment: lane = column N (caller adds lane&15),
// half selects K in {h*8..h*8+7} and {16+h*8..16+h*8+7}. src is B^T row-major (ld = K dim).
__device__ inline v16bf load_frag_b(const bf16_t* src, int col, int kbase, int ld) {
  const int lane = threadIdx.x & 31;
  const int half = lane >> 4;
  const bf16_t* p = src + (size_t)col * ld + kbase + half * 8;
  v8bf lo = *(const v8bf*)p;
  v8bf hi = *(const v8bf*)(p + 16);
  return __builtin_shufflevector(lo, hi, 0,1,2,3,4,5,6,7,8,9,10,11,12,13,14,15);
}

__device__ inline v8f wmma_bf16(v16bf a, v16bf b, v8f c) {
  return __builtin_amdgcn_wmma_f32_16x16x32_bf16(false, a, false, b, (short)0, c, false, false);
}

__device__ inline float rmax16(float v) {
  v = fmaxf(v, __shfl_xor(v, 1, 32));
  v = fmaxf(v, __shfl_xor(v, 2, 32));
  v = fmaxf(v, __shfl_xor(v, 4, 32));
  v = fmaxf(v, __shfl_xor(v, 8, 32));
  return v;
}
__device__ inline float rsum16(float v) {
  v += __shfl_xor(v, 1, 32);
  v += __shfl_xor(v, 2, 32);
  v += __shfl_xor(v, 4, 32);
  v += __shfl_xor(v, 8, 32);
  return v;
}

// ---------- conversion kernels ----------

__global__ void cvt_bf16_kernel(const float* __restrict__ src, bf16_t* __restrict__ dst, int n) {
  for (int i = blockIdx.x * blockDim.x + threadIdx.x; i < n; i += gridDim.x * blockDim.x)
    dst[i] = f2bf(src[i]);
}

// src [rows][cols] f32 -> dst [cols][rows] bf16
__global__ void cvt_transpose_kernel(const float* __restrict__ src, bf16_t* __restrict__ dst,
                                     int rows, int cols) {
  int n = rows * cols;
  for (int i = blockIdx.x * blockDim.x + threadIdx.x; i < n; i += gridDim.x * blockDim.x) {
    int r = i / cols, c = i - r * cols;
    dst[(size_t)c * rows + r] = f2bf(src[i]);
  }
}

// ---------- software-pipelined 32x64-per-wave GEMM core ----------
// A row-major [M,1024] bf16, Bt = B^T row-major [N,1024] bf16.
// acc[sub][nt] covers rows mb+sub*16.., cols nb+nt*16..
struct Frags {
  v16bf a0, a1, b0, b1, b2, b3;
};
__device__ inline Frags load_frags(const bf16_t* A, const bf16_t* Bt,
                                   int mb, int nb, int kb, int r) {
  Frags f;
  f.a0 = load_frag_a(A, mb + r,      kb, 1024);
  f.a1 = load_frag_a(A, mb + 16 + r, kb, 1024);
  f.b0 = load_frag_b(Bt, nb + r,      kb, 1024);
  f.b1 = load_frag_b(Bt, nb + 16 + r, kb, 1024);
  f.b2 = load_frag_b(Bt, nb + 32 + r, kb, 1024);
  f.b3 = load_frag_b(Bt, nb + 48 + r, kb, 1024);
  return f;
}

__device__ inline void gemm_core(const bf16_t* __restrict__ A, const bf16_t* __restrict__ Bt,
                                 int mb, int nb, int r, v8f acc[2][4]) {
#pragma unroll
  for (int s = 0; s < 2; s++)
#pragma unroll
    for (int nt = 0; nt < 4; nt++) acc[s][nt] = V8F_ZERO;

  Frags cur = load_frags(A, Bt, mb, nb, 0, r);
  for (int kb = 0; kb < 1024; kb += 32) {
    Frags nxt;
    if (kb + 32 < 1024) {
      __builtin_prefetch(A + (size_t)(mb + r) * 1024 + kb + 96, 0, 1);
      nxt = load_frags(A, Bt, mb, nb, kb + 32, r);   // in flight under the WMMAs
    }
    acc[0][0] = wmma_bf16(cur.a0, cur.b0, acc[0][0]);
    acc[0][1] = wmma_bf16(cur.a0, cur.b1, acc[0][1]);
    acc[0][2] = wmma_bf16(cur.a0, cur.b2, acc[0][2]);
    acc[0][3] = wmma_bf16(cur.a0, cur.b3, acc[0][3]);
    acc[1][0] = wmma_bf16(cur.a1, cur.b0, acc[1][0]);
    acc[1][1] = wmma_bf16(cur.a1, cur.b1, acc[1][1]);
    acc[1][2] = wmma_bf16(cur.a1, cur.b2, acc[1][2]);
    acc[1][3] = wmma_bf16(cur.a1, cur.b3, acc[1][3]);
    cur = nxt;
  }
}

// ---------- QKV projection GEMM:  [8192,1024] x [1024,3072] ----------
// Epilogue scatters bf16 into Q[B,H,L,64], K[B,H,L,64], Vt[B,H,64,L].
__global__ __launch_bounds__(128) void gemm_qkv_kernel(
    const bf16_t* __restrict__ A, const bf16_t* __restrict__ Bt,
    bf16_t* __restrict__ Qo, bf16_t* __restrict__ Ko, bf16_t* __restrict__ Vto) {
  const int lane = threadIdx.x & 31, w = threadIdx.x >> 5;
  const int half = lane >> 4, r = lane & 15;
  const int mb = blockIdx.x * 128 + w * 32;
  const int nb = blockIdx.y * 64;

  v8f acc[2][4];
  gemm_core(A, Bt, mb, nb, r, acc);

  const int t = nb >> 10;             // 0=Q 1=K 2=V (columns = t*1024 + h*64 + hd)
  const int h = (nb & 1023) >> 6;     // constant per block column (64-aligned)
#pragma unroll
  for (int s = 0; s < 2; s++) {
#pragma unroll
    for (int nt = 0; nt < 4; nt++) {
#pragma unroll
      for (int i = 0; i < 8; i++) {
        int m  = mb + s * 16 + i + half * 8;   // global row in [0, 8192)
        int hd = nt * 16 + r;                  // 0..63
        int bb = m >> 11, lp = m & 2047;
        bf16_t v = f2bf(acc[s][nt][i]);
        size_t idx = (((size_t)(bb * 16 + h)) * 2048 + lp) * 64 + hd;
        if (t == 0)      Qo[idx] = v;
        else if (t == 1) Ko[idx] = v;
        else             Vto[(((size_t)(bb * 16 + h)) * 64 + hd) * 2048 + lp] = v;
      }
    }
  }
}

// ---------- flash attention: one wave per 16-row Q tile, pipelined K loads ----------
__global__ __launch_bounds__(128) void attn_kernel(
    const bf16_t* __restrict__ Q, const bf16_t* __restrict__ K,
    const bf16_t* __restrict__ Vt, bf16_t* __restrict__ Aout) {
  __shared__ __align__(16) bf16_t lds_p[4][16][32];

  const int lane = threadIdx.x & 31, w = threadIdx.x >> 5;
  const int half = lane >> 4, r = lane & 15;
  const int tile = blockIdx.x * 4 + w;      // 8192 tiles
  const int qt = tile & 127;                // L/16
  const int bh = tile >> 7;                 // b*16 + h
  const int qb = qt * 16;

  const bf16_t* Qp = Q  + (size_t)bh * 2048 * 64;
  const bf16_t* Kp = K  + (size_t)bh * 2048 * 64;
  const bf16_t* Vp = Vt + (size_t)bh * 64 * 2048;

  v16bf q0 = load_frag_a(Qp, qb + r, 0,  64);
  v16bf q1 = load_frag_a(Qp, qb + r, 32, 64);

  v8f acc[4];
#pragma unroll
  for (int nt = 0; nt < 4; nt++) acc[nt] = V8F_ZERO;
  float mstat[8], lstat[8];
#pragma unroll
  for (int i = 0; i < 8; i++) { mstat[i] = -1e30f; lstat[i] = 0.f; }

  const float scale = 0.125f;               // 1/sqrt(64)
  const int kend = qb + 16;                 // causal: keys < kend

  // prologue: K fragments for first key chunk
  v16bf kf0 = load_frag_b(Kp, r,      0,  64);
  v16bf kf1 = load_frag_b(Kp, r,      32, 64);
  v16bf kf2 = load_frag_b(Kp, 16 + r, 0,  64);
  v16bf kf3 = load_frag_b(Kp, 16 + r, 32, 64);

  for (int kb = 0; kb < kend; kb += 32) {
    // S tile: Q(16x64) x K^T for keys kb..kb+31 (fragments already in flight)
    v8f s0 = V8F_ZERO, s1 = V8F_ZERO;
    s0 = wmma_bf16(q0, kf0, s0);
    s0 = wmma_bf16(q1, kf1, s0);
    s1 = wmma_bf16(q0, kf2, s1);
    s1 = wmma_bf16(q1, kf3, s1);

    // issue next chunk's K fragments + this chunk's V fragments now; they fly
    // under the softmax VALU work and the LDS round-trip below.
    if (kb + 32 < kend) {
      kf0 = load_frag_b(Kp, kb + 32 + r, 0,  64);
      kf1 = load_frag_b(Kp, kb + 32 + r, 32, 64);
      kf2 = load_frag_b(Kp, kb + 48 + r, 0,  64);
      kf3 = load_frag_b(Kp, kb + 48 + r, 32, 64);
    }
    v16bf vf0 = load_frag_b(Vp, r,      kb, 2048);   // V^T: contiguous keys
    v16bf vf1 = load_frag_b(Vp, 16 + r, kb, 2048);
    v16bf vf2 = load_frag_b(Vp, 32 + r, kb, 2048);
    v16bf vf3 = load_frag_b(Vp, 48 + r, kb, 2048);

    // online softmax, per accumulator row (row = qb + i + 8*half, col = kb(+16) + r)
#pragma unroll
    for (int i = 0; i < 8; i++) {
      int row = qb + i + half * 8;
      float v0 = s0[i] * scale;
      float v1 = s1[i] * scale;
      if (kb + r > row)      v0 = -1e30f;
      if (kb + 16 + r > row) v1 = -1e30f;
      float mnew = fmaxf(mstat[i], rmax16(fmaxf(v0, v1)));
      float p0 = __expf(v0 - mnew);
      float p1 = __expf(v1 - mnew);
      float rs = rsum16(p0 + p1);
      float corr = __expf(mstat[i] - mnew);
      lstat[i] = lstat[i] * corr + rs;
      mstat[i] = mnew;
      acc[0][i] *= corr; acc[1][i] *= corr; acc[2][i] *= corr; acc[3][i] *= corr;
      lds_p[w][i + half * 8][r]      = f2bf(p0);
      lds_p[w][i + half * 8][16 + r] = f2bf(p1);
    }
    // same-wave DS ordering; make it explicit and block compiler reordering
    asm volatile("s_wait_dscnt 0x0" ::: "memory");
    v16bf pf = load_frag_a(&lds_p[w][0][0], r, 0, 32);
    acc[0] = wmma_bf16(pf, vf0, acc[0]);
    acc[1] = wmma_bf16(pf, vf1, acc[1]);
    acc[2] = wmma_bf16(pf, vf2, acc[2]);
    acc[3] = wmma_bf16(pf, vf3, acc[3]);
  }

  // epilogue: O/l -> attn_bf16 [B, L, H*64] (== [8192,1024] row-major)
  const int b = bh >> 4, h = bh & 15;
#pragma unroll
  for (int i = 0; i < 8; i++) {
    float inv = 1.0f / lstat[i];
    int row = qb + i + half * 8;
    size_t base = ((size_t)b * 2048 + row) * 1024 + h * 64;
#pragma unroll
    for (int nt = 0; nt < 4; nt++)
      Aout[base + nt * 16 + r] = f2bf(acc[nt][i] * inv);
  }
}

// ---------- output projection: [8192,1024] x [1024,1024] -> f32 ----------
__global__ __launch_bounds__(128) void gemm_proj_kernel(
    const bf16_t* __restrict__ A, const bf16_t* __restrict__ Bt,
    float* __restrict__ out) {
  const int lane = threadIdx.x & 31, w = threadIdx.x >> 5;
  const int half = lane >> 4, r = lane & 15;
  const int mb = blockIdx.x * 128 + w * 32;
  const int nb = blockIdx.y * 64;

  v8f acc[2][4];
  gemm_core(A, Bt, mb, nb, r, acc);

#pragma unroll
  for (int s = 0; s < 2; s++)
#pragma unroll
    for (int nt = 0; nt < 4; nt++)
#pragma unroll
      for (int i = 0; i < 8; i++) {
        int m = mb + s * 16 + i + half * 8;
        out[(size_t)m * 1024 + nb + nt * 16 + r] = acc[s][nt][i];
      }
}

// ---------- host ----------

extern "C" void kernel_launch(void* const* d_in, const int* in_sizes, int n_in,
                              void* d_out, int out_size, void* d_ws, size_t ws_size,
                              hipStream_t stream) {
  const float* z     = (const float*)d_in[0];   // [4,2048,1024]
  const float* Wqkv  = (const float*)d_in[1];   // [1024,3072]
  const float* Wproj = (const float*)d_in[2];   // [1024,1024]
  float* out = (float*)d_out;                   // [4,2048,1024] f32

  char* ws = (char*)d_ws;
  size_t off = 0;
  auto carve = [&](size_t bytes) -> char* {
    char* p = ws + off;
    off += (bytes + 255) & ~(size_t)255;
    return p;
  };
  bf16_t* zb     = (bf16_t*)carve(8192ull * 1024 * 2);
  bf16_t* wqkvT  = (bf16_t*)carve(3072ull * 1024 * 2);
  bf16_t* wprojT = (bf16_t*)carve(1024ull * 1024 * 2);
  bf16_t* Qw     = (bf16_t*)carve(64ull * 2048 * 64 * 2);
  bf16_t* Kw     = (bf16_t*)carve(64ull * 2048 * 64 * 2);
  bf16_t* Vtw    = (bf16_t*)carve(64ull * 64 * 2048 * 2);
  bf16_t* attn   = (bf16_t*)carve(8192ull * 1024 * 2);

  cvt_bf16_kernel<<<2048, 256, 0, stream>>>(z, zb, 8192 * 1024);
  cvt_transpose_kernel<<<2048, 256, 0, stream>>>(Wqkv, wqkvT, 1024, 3072);
  cvt_transpose_kernel<<<1024, 256, 0, stream>>>(Wproj, wprojT, 1024, 1024);

  gemm_qkv_kernel<<<dim3(64, 48), 128, 0, stream>>>(zb, wqkvT, Qw, Kw, Vtw);
  attn_kernel<<<2048, 128, 0, stream>>>(Qw, Kw, Vtw, attn);
  gemm_proj_kernel<<<dim3(64, 16), 128, 0, stream>>>(attn, wprojT, out);
}
